// Linear_19198503813358
// MI455X (gfx1250) — compile-verified, matrix operand order checked by reading, not run on hardware
//
#include <hip/hip_runtime.h>
#include <stdint.h>

typedef __attribute__((ext_vector_type(16))) _Float16 v16h;
typedef __attribute__((ext_vector_type(8)))  float    v8f;
typedef __attribute__((ext_vector_type(4)))  int      v4i;

// Problem dims (fixed by the reference): M = B*S, N = D_OUT, K = D_IN
#define MDIM 8192
#define NDIM 4096
#define KDIM 4096

// GEMM tiling: block = 256x128, 8 waves in 4(M) x 2(N), wave tile 64x64
#define BM 256
#define BN 128
#define BK 32   // one V_WMMA_F32_16X16X32_F16 K-step

// ---- CDNA5 async global->LDS DMA (signature probe-confirmed in round 2:
//      (int4 AS1*, int4 AS3*, imm int, imm int)) ---------------------------
#if defined(__has_builtin)
#if __has_builtin(__builtin_amdgcn_global_load_async_to_lds_b128)
#define HAVE_ASYNC_LDS 1
#endif
#endif

#define AS_LDS_V4I(p) ((__attribute__((address_space(3))) v4i*)(uintptr_t)(p))
#define AS_GLB_V4I(p) ((__attribute__((address_space(1))) v4i*)(uintptr_t)(p))

__device__ __forceinline__ void wait_async_zero() {
#if defined(__has_builtin)
#if __has_builtin(__builtin_amdgcn_s_wait_asynccnt)
    __builtin_amdgcn_s_wait_asynccnt(0);
#else
    asm volatile("s_wait_asynccnt 0" ::: "memory");
#endif
#endif
}

// ---------------------------------------------------------------------------
// Phase 1a: dequantize int8 weight * fp32 per-element scale -> f16, stored
// K-major (transposed: Wt[n * K + k]) so GEMM B-tiles are contiguous in K.
// 32x32 tile transpose through LDS for coalesced reads AND writes.
// ---------------------------------------------------------------------------
__global__ __launch_bounds__(256)
void dequant_transpose_kernel(const int8_t* __restrict__ w,
                              const float*  __restrict__ wsc,
                              _Float16*     __restrict__ wt) {
    __shared__ _Float16 tile[32][33];   // +1 pad: conflict-free transposed read
    const int n0 = blockIdx.x * 32;
    const int k0 = blockIdx.y * 32;
    const int c  = threadIdx.x & 31;
    const int rg = threadIdx.x >> 5;    // 0..7

#pragma unroll
    for (int i = 0; i < 4; ++i) {
        const int r = rg + i * 8;                       // k within tile
        const size_t g = (size_t)(k0 + r) * NDIM + (n0 + c);
        tile[r][c] = (_Float16)((float)w[g] * wsc[g]);
    }
    __syncthreads();
#pragma unroll
    for (int i = 0; i < 4; ++i) {
        const int rn = rg + i * 8;                      // n within tile
        wt[(size_t)(n0 + rn) * KDIM + (k0 + c)] = tile[c][rn];
    }
}

// ---------------------------------------------------------------------------
// Phase 1b (optional, used when workspace allows): convert A fp32 -> f16 once
// so the GEMM's A staging becomes a pure async global->LDS DMA.
// 16 elements per thread: 64B load, 32B store, fully coalesced.
// ---------------------------------------------------------------------------
__global__ __launch_bounds__(256)
void convertA_kernel(const float* __restrict__ A, _Float16* __restrict__ A16) {
    const size_t i = ((size_t)blockIdx.x * 256 + threadIdx.x) * 16;
    float4 f[4];
#pragma unroll
    for (int j = 0; j < 4; ++j) f[j] = ((const float4*)(A + i))[j];
    const float* fs = (const float*)f;
    union { _Float16 h[16]; uint4 u[2]; } t;
#pragma unroll
    for (int j = 0; j < 16; ++j) t.h[j] = (_Float16)fs[j];
    ((uint4*)(A16 + i))[0] = t.u[0];
    ((uint4*)(A16 + i))[1] = t.u[1];
}

// ---------------------------------------------------------------------------
// WMMA fragment load from an LDS tile stored with K-stride BK (=32 halves).
// 16-bit A (16x32) layout per CDNA5 ISA:
//   lane 0..15  : row = lane,    halves 0..7 = K 0..7,  halves 8..15 = K 16..23
//   lane 16..31 : row = lane-16, halves 0..7 = K 8..15, halves 8..15 = K 24..31
// Two 16-byte DS loads per lane.
// ---------------------------------------------------------------------------
__device__ __forceinline__ v16h load_frag(const _Float16* t, int row0, int lane) {
    const int r    = row0 + (lane & 15);
    const int koff = (lane & 16) ? 8 : 0;
    const uint4* p = (const uint4*)(t + r * BK + koff);
    union { uint4 u[2]; v16h h; } f;
    f.u[0] = p[0];   // K koff   .. koff+7
    f.u[1] = p[2];   // K koff+16.. koff+23
    return f.h;
}

// ---------------------------------------------------------------------------
// Phase 2: GEMM.  grid = (N/BN, M/BM), 256 threads = 8 waves (4 M x 2 N).
// Each wave: 64x64 output = 4x4 accumulators of 16x16 fp32 (16 WMMA / K-step).
// AF16=true : A pre-converted -> A staging is async DMA (no VALU, no VGPR trip)
// AF16=false: A converted fp32->f16 during staging (fallback, small ws)
// B always staged via async DMA.  Double-buffered LDS, 1 barrier per K-step.
// B-fragments software-pipelined one quad ahead to hide LDS latency.
// ---------------------------------------------------------------------------
template <bool AF16>
__global__ __launch_bounds__(256, 1)
void gemm_f16_wmma_kernel(const float*    __restrict__ A32,
                          const _Float16* __restrict__ A16,  // [M, K] f16 (AF16)
                          const _Float16* __restrict__ Bt,   // [N, K] K-major f16
                          const float*    __restrict__ bias,
                          float*          __restrict__ out) {
    __shared__ _Float16 As[2][BM * BK];   // 2 x 16 KB
    __shared__ _Float16 Bs[2][BN * BK];   // 2 x  8 KB

    const int n0   = blockIdx.x * BN;
    const int m0   = blockIdx.y * BM;
    const int tid  = threadIdx.x;
    const int lane = tid & 31;
    const int wave = tid >> 5;
    const int wm   = wave >> 1;   // 0..3  (64 rows each)
    const int wn   = wave & 1;    // 0..1  (64 cols each)

    v8f acc[4][4];
    const v8f vzero = {0.f, 0.f, 0.f, 0.f, 0.f, 0.f, 0.f, 0.f};
#pragma unroll
    for (int a = 0; a < 4; ++a)
#pragma unroll
        for (int b = 0; b < 4; ++b) acc[a][b] = vzero;

    // A staging: 1 thread per row (32 K values = 64B of f16 per row).
    auto stageA = [&](int kk, int pb) {
        if constexpr (AF16) {
            const _Float16* src = A16 + (size_t)(m0 + tid) * KDIM + kk;
            _Float16*       dst = &As[pb][tid * BK];
#if defined(HAVE_ASYNC_LDS)
#pragma unroll
            for (int j = 0; j < 4; ++j)
                __builtin_amdgcn_global_load_async_to_lds_b128(
                    AS_GLB_V4I(src + j * 8), AS_LDS_V4I(dst + j * 8), 0, 0);
#else
            const uint4* s = (const uint4*)src;
            uint4*       d = (uint4*)dst;
#pragma unroll
            for (int j = 0; j < 4; ++j) d[j] = s[j];
#endif
        } else {
            const float* ag = A32 + (size_t)(m0 + tid) * KDIM + kk;
            float4 f[8];
#pragma unroll
            for (int j = 0; j < 8; ++j) f[j] = ((const float4*)ag)[j];
            const float* fs = (const float*)f;
            union { _Float16 h[32]; uint4 u[4]; } t;
#pragma unroll
            for (int j = 0; j < 32; ++j) t.h[j] = (_Float16)fs[j];
            uint4* dst = (uint4*)&As[pb][tid * BK];
#pragma unroll
            for (int j = 0; j < 4; ++j) dst[j] = t.u[j];
        }
    };

    // B staging: 2 threads per n-row, 32B (16 f16) each; pure async DMA copy.
    auto stageB = [&](int kk, int pb) {
        const int row  = tid >> 1;
        const int half = tid & 1;
        const _Float16* src = Bt + (size_t)(n0 + row) * KDIM + kk + half * 16;
        _Float16*       dst = &Bs[pb][row * BK + half * 16];
#if defined(HAVE_ASYNC_LDS)
        __builtin_amdgcn_global_load_async_to_lds_b128(AS_GLB_V4I(src),     AS_LDS_V4I(dst),     0, 0);
        __builtin_amdgcn_global_load_async_to_lds_b128(AS_GLB_V4I(src + 8), AS_LDS_V4I(dst + 8), 0, 0);
#else
        const uint4* s = (const uint4*)src;
        uint4*       d = (uint4*)dst;
        d[0] = s[0];
        d[1] = s[1];
#endif
    };

    stageA(0, 0);
    stageB(0, 0);
    wait_async_zero();
    __syncthreads();

    int pb = 0;
    for (int kk = 0; kk < KDIM; kk += BK, pb ^= 1) {
        if (kk + BK < KDIM) {
            stageA(kk + BK, pb ^ 1);
            stageB(kk + BK, pb ^ 1);
        }

        const _Float16* at = As[pb];
        const _Float16* bt = Bs[pb];

        // A-frags: load all 4 up front (reused by every quad).
        v16h afr[4];
#pragma unroll
        for (int a = 0; a < 4; ++a)
            afr[a] = load_frag(at, wm * 64 + a * 16, lane);

        // B-frags: software-pipelined one quad ahead (2 live at a time) so the
        // DS loads for quad b+1 are in flight while quad b's WMMAs execute.
        v16h bcur = load_frag(bt, wn * 64 + 0, lane);
#pragma unroll
        for (int b = 0; b < 4; ++b) {
            v16h bnext;
            if (b < 3) bnext = load_frag(bt, wn * 64 + (b + 1) * 16, lane);
#pragma unroll
            for (int a = 0; a < 4; ++a)
                acc[a][b] = __builtin_amdgcn_wmma_f32_16x16x32_f16(
                    false, afr[a], false, bcur,
                    (short)0, acc[a][b], false, false);
            bcur = bnext;
        }

        wait_async_zero();   // next-buffer async tiles must have landed in LDS
        __syncthreads();
    }

    // ---- epilogue: C layout (32-bit 16x16): lanes 0-15 -> N=lane, M=r;
    //               lanes 16-31 -> N=lane-16, M=r+8.  Add bias, store fp32.
    const int cn   = lane & 15;
    const int mofs = (lane & 16) ? 8 : 0;
#pragma unroll
    for (int a = 0; a < 4; ++a) {
#pragma unroll
        for (int b = 0; b < 4; ++b) {
            const int gcol = n0 + wn * 64 + b * 16 + cn;
            const float bv = bias[gcol];
            const int grow = m0 + wm * 64 + a * 16 + mofs;
            float* op = out + (size_t)grow * NDIM + gcol;
#pragma unroll
            for (int r = 0; r < 8; ++r)
                op[(size_t)r * NDIM] = acc[a][b][r] + bv;
        }
    }
}

// ---------------------------------------------------------------------------
// Launcher
// ---------------------------------------------------------------------------
extern "C" void kernel_launch(void* const* d_in, const int* in_sizes, int n_in,
                              void* d_out, int out_size, void* d_ws, size_t ws_size,
                              hipStream_t stream) {
    (void)in_sizes; (void)n_in; (void)out_size;

    const float*  A    = (const float*)d_in[0];     // [B,S,D_IN] fp32
    const int8_t* w    = (const int8_t*)d_in[1];    // [D_IN,D_OUT] int8
    const float*  wsc  = (const float*)d_in[2];     // [D_IN,D_OUT] fp32
    const float*  bias = (const float*)d_in[3];     // [D_OUT] fp32
    float*        out  = (float*)d_out;             // [B,S,D_OUT] fp32

    const size_t W16_BYTES = (size_t)NDIM * KDIM * sizeof(_Float16); // 32 MiB
    const size_t A16_BYTES = (size_t)MDIM * KDIM * sizeof(_Float16); // 64 MiB

    _Float16* wt = (_Float16*)d_ws;

    dequant_transpose_kernel<<<dim3(NDIM / 32, KDIM / 32), 256, 0, stream>>>(w, wsc, wt);

    if (ws_size >= W16_BYTES + A16_BYTES) {
        _Float16* a16 = (_Float16*)((char*)d_ws + W16_BYTES);
        convertA_kernel<<<(MDIM * (size_t)KDIM) / (16 * 256), 256, 0, stream>>>(A, a16);
        gemm_f16_wmma_kernel<true><<<dim3(NDIM / BN, MDIM / BM), 256, 0, stream>>>(
            A, a16, wt, bias, out);
    } else {
        gemm_f16_wmma_kernel<false><<<dim3(NDIM / BN, MDIM / BM), 256, 0, stream>>>(
            A, (const _Float16*)nullptr, wt, bias, out);
    }
}
